// CTRNN_76338748719531
// MI455X (gfx1250) — compile-verified, matrix operand order checked by reading, not run on hardware
//
#include <hip/hip_runtime.h>
#include <hip/hip_bf16.h>

typedef __bf16 v16bf __attribute__((ext_vector_type(16)));
typedef __bf16 v8bf  __attribute__((ext_vector_type(8)));
typedef float  v8f   __attribute__((ext_vector_type(8)));

#define TILE_M 128
#define TILE_N 128
#define KSTEP 32
#define LDS_STRIDE 40                             // bf16 elems; 80B padded rows
#define LDS_BUF_BYTES (TILE_M * LDS_STRIDE * 2)   // 10240 B per buffer

__device__ __forceinline__ unsigned cvt_pk_bf16(float a, float b) {
  unsigned r;
  asm("v_cvt_pk_bf16_f32 %0, %1, %2" : "=v"(r) : "v"(a), "v"(b));
  return r;
}

__device__ __forceinline__ float tanh_hw(float x) {
  float r;
  asm("v_tanh_f32 %0, %1\n\tv_nop" : "=v"(r) : "v"(x));
  return r;
}

// Async global->LDS 16B copy per lane (GVS: SGPR base + 32b lane offset). ASYNCcnt.
__device__ __forceinline__ void async_b128(unsigned lds_addr, unsigned voff, const void* base) {
  asm volatile("global_load_async_to_lds_b128 %0, %1, %2"
               :: "v"(lds_addr), "v"(voff), "s"(base) : "memory");
}
__device__ __forceinline__ void wait_async0() {
  asm volatile("s_wait_asynccnt 0x0" ::: "memory");
}

// One-time prep: strided f32 -> packed bf16 pairs.
__global__ __launch_bounds__(256)
void cvt_f32_to_bf16(const float* __restrict__ src, int ld, int cols,
                     unsigned* __restrict__ dst, long long npairs) {
  long long i = (long long)blockIdx.x * blockDim.x + threadIdx.x;
  if (i >= npairs) return;
  int pc = cols >> 1;
  long long row = i / pc;
  int c2 = (int)(i - row * pc) << 1;
  const float* p = src + row * (long long)ld + c2;
  dst[i] = cvt_pk_bf16(p[0], p[1]);
}

// mode 0:  out0 = A*W^T + bias                      (x_bf16 -> xw f32)
// mode 1:  hn = 0.9*hold + 0.1*tanh(xw + A*W^T)
//          out0 = hn, out1 = dup (final), outbf = bf16(hn) (next unfold's A)
__global__ __launch_bounds__(256)
void ctrnn_gemm(const unsigned short* __restrict__ Abf, int lda,
                const unsigned short* __restrict__ Wbf, int ldw, int K,
                const float* __restrict__ bias,
                const float* __restrict__ xw,
                const float* __restrict__ hold,
                float* __restrict__ out0,
                float* __restrict__ out1,
                unsigned short* __restrict__ outbf,
                int mode)
{
  __shared__ unsigned short ldsA[2][TILE_M * LDS_STRIDE];
  __shared__ unsigned short ldsB[2][TILE_N * LDS_STRIDE];

  const int tid   = threadIdx.x;
  const int lane  = tid & 31;
  const int wave  = tid >> 5;                 // 0..7
  const int mBase = blockIdx.y * TILE_M;
  const int nBase = blockIdx.x * TILE_N;

  // Async fill: each thread owns 16B chunks at (row r0, sub) and (row r0+64, sub).
  const int r0  = tid >> 2;                   // 0..63
  const int sub = tid & 3;                    // 0..3
  const unsigned short* aBase = Abf + (size_t)mBase * lda;
  const unsigned short* wBase = Wbf + (size_t)nBase * ldw;
  const unsigned aoff0 = (unsigned)(r0 * lda * 2 + sub * 16);
  const unsigned aoff1 = aoff0 + (unsigned)(64 * lda * 2);
  const unsigned woff0 = (unsigned)(r0 * ldw * 2 + sub * 16);
  const unsigned woff1 = woff0 + (unsigned)(64 * ldw * 2);
  const unsigned ldsAoff = (unsigned)(size_t)&ldsA[0][0] + (unsigned)(r0 * 80 + sub * 16);
  const unsigned ldsBoff = (unsigned)(size_t)&ldsB[0][0] + (unsigned)(r0 * 80 + sub * 16);

  v8f acc[8] = {};
  const int klo = (lane & 16) ? 8 : 0;        // ISA bf16 fragment K split
  const int l15 = lane & 15;

  // Compute one staged K tile: 1 A fragment + 8 pipelined B fragments + 8 WMMAs.
  auto compute_tile = [&](int cb) {
    int arow = wave * 16 + l15;
    v8bf alo = *(const v8bf*)&ldsA[cb][arow * LDS_STRIDE + klo];
    v8bf ahi = *(const v8bf*)&ldsA[cb][arow * LDS_STRIDE + klo + 16];
    v16bf afrag;
#pragma unroll
    for (int j = 0; j < 8; ++j) { afrag[j] = alo[j]; afrag[j + 8] = ahi[j]; }

    v8bf blo[2], bhi[2];
    blo[0] = *(const v8bf*)&ldsB[cb][l15 * LDS_STRIDE + klo];
    bhi[0] = *(const v8bf*)&ldsB[cb][l15 * LDS_STRIDE + klo + 16];
#pragma unroll
    for (int t = 0; t < 8; ++t) {
      if (t + 1 < 8) {                        // prefetch next B fragment (alt regs)
        int brow = (t + 1) * 16 + l15;
        blo[(t + 1) & 1] = *(const v8bf*)&ldsB[cb][brow * LDS_STRIDE + klo];
        bhi[(t + 1) & 1] = *(const v8bf*)&ldsB[cb][brow * LDS_STRIDE + klo + 16];
      }
      v16bf bfrag;
#pragma unroll
      for (int j = 0; j < 8; ++j) { bfrag[j] = blo[t & 1][j]; bfrag[j + 8] = bhi[t & 1][j]; }
      acc[t] = __builtin_amdgcn_wmma_f32_16x16x32_bf16(
          false, afrag, false, bfrag, (short)0, acc[t], false, false);
    }
  };

  // ---- prologue: async-stage first K tile into buffer 0 ----
  async_b128(ldsAoff,           aoff0, aBase);
  async_b128(ldsAoff + 64 * 80, aoff1, aBase);
  async_b128(ldsBoff,           woff0, wBase);
  async_b128(ldsBoff + 64 * 80, woff1, wBase);
  wait_async0();
  __syncthreads();

  // ---- steady state: peel the last step; no conditionals inside ----
  const int nsteps = K / KSTEP;
  int cur = 0;
  for (int s = 0; s < nsteps - 1; ++s) {
    unsigned kb = (unsigned)((s + 1) * (KSTEP * 2));
    unsigned la = ldsAoff + (cur ^ 1) * LDS_BUF_BYTES;
    unsigned lb = ldsBoff + (cur ^ 1) * LDS_BUF_BYTES;
    async_b128(la,           aoff0 + kb, aBase);
    async_b128(la + 64 * 80, aoff1 + kb, aBase);
    async_b128(lb,           woff0 + kb, wBase);
    async_b128(lb + 64 * 80, woff1 + kb, wBase);

    compute_tile(cur);

    wait_async0();
    __syncthreads();
    cur ^= 1;
  }
  compute_tile(cur);                           // last K tile

  // ---- epilogue: lane l%16 = col; vgpr r + 8*(lane/16) = row ----
  const int mlo = (lane & 16) ? 8 : 0;
  const int N = 1024;
  if (mode == 0) {
#pragma unroll
    for (int t = 0; t < 8; ++t) {
      int col = nBase + t * 16 + l15;
      float bv = bias[col];
#pragma unroll
      for (int r = 0; r < 8; ++r) {
        int row = mBase + wave * 16 + mlo + r;
        out0[(size_t)row * N + col] = acc[t][r] + bv;
      }
    }
  } else {
#pragma unroll
    for (int t = 0; t < 8; ++t) {
      int col = nBase + t * 16 + l15;
#pragma unroll
      for (int r = 0; r < 8; ++r) {
        int row = mBase + wave * 16 + mlo + r;
        size_t idx = (size_t)row * N + col;
        float s  = xw[idx] + acc[t][r];
        float f  = tanh_hw(s);
        float hn = 0.9f * hold[idx] + 0.1f * f;   // h + dt*(-h/tau + f)
        out0[idx] = hn;
        if (out1)  out1[idx] = hn;
        if (outbf) outbf[idx] = (unsigned short)cvt_pk_bf16(hn, hn);
      }
    }
  }
}

extern "C" void kernel_launch(void* const* d_in, const int* in_sizes, int n_in,
                              void* d_out, int out_size, void* d_ws, size_t ws_size,
                              hipStream_t stream) {
  const float* x    = (const float*)d_in[0];   // 16384 x 512
  const float* h    = (const float*)d_in[1];   // 16384 x 1024
  const float* fc_w = (const float*)d_in[2];   // 1024 x 1536 (row-major)
  const float* fc_b = (const float*)d_in[3];   // 1024
  float* out = (float*)d_out;                  // (h, h) concatenated

  const size_t BH = (size_t)16384 * 1024;

  // Workspace: xw(64M) hf(64M) hbfA(32M) hbfB(32M) xbf(16M) wxbf(1M) whbf(2M)
  float* xw = (float*)d_ws;
  float* hf = xw + BH;
  unsigned short* hbfA = (unsigned short*)(hf + BH);
  unsigned short* hbfB = hbfA + BH;
  unsigned short* xbf  = hbfB + BH;
  unsigned short* wxbf = xbf + (size_t)16384 * 512;
  unsigned short* whbf = wxbf + (size_t)1024 * 512;

  // ---- one-time f32 -> bf16 prep ----
  {
    long long np;
    np = 16384LL * 256;
    cvt_f32_to_bf16<<<(np + 255) / 256, 256, 0, stream>>>(x, 512, 512, (unsigned*)xbf, np);
    np = 16384LL * 512;
    cvt_f32_to_bf16<<<(np + 255) / 256, 256, 0, stream>>>(h, 1024, 1024, (unsigned*)hbfA, np);
    np = 1024LL * 256;
    cvt_f32_to_bf16<<<(np + 255) / 256, 256, 0, stream>>>(fc_w, 1536, 512, (unsigned*)wxbf, np);
    np = 1024LL * 512;
    cvt_f32_to_bf16<<<(np + 255) / 256, 256, 0, stream>>>(fc_w + 512, 1536, 1024, (unsigned*)whbf, np);
  }

  dim3 grid(1024 / TILE_N, 16384 / TILE_M);    // (8, 128)

  // GEMM1: xw = x * w_x^T + b
  ctrnn_gemm<<<grid, 256, 0, stream>>>(xbf, 512, wxbf, 512, 512,
                                       fc_b, nullptr, nullptr, xw, nullptr, nullptr, 0);

  // 6 unfolds: h = 0.9h + 0.1*tanh(xw + h * w_h^T)
  unsigned short* bf_bufs[2] = { hbfA, hbfB };
  const float* hold = h;
  for (int it = 0; it < 6; ++it) {
    const unsigned short* inbf = bf_bufs[it & 1];
    unsigned short* obf = (it == 5) ? nullptr : bf_bufs[(it + 1) & 1];
    float* o0;
    float* o1 = nullptr;
    if (it == 5) { o0 = out; o1 = out + BH; }
    else         { o0 = hf; }                  // f32 h updated in place (elementwise only)
    ctrnn_gemm<<<grid, 256, 0, stream>>>(inbf, 1024, whbf, 1024, 1024,
                                         nullptr, xw, hold, o0, o1, obf, 1);
    hold = o0;
  }
}
